// SpatioTemporalTransformerBlock_4750233829700
// MI455X (gfx1250) — matrix operand order repaired and checked by reading, not compiled, not run on hardware
//
#include <hip/hip_runtime.h>
#include <math.h>
#include <stdint.h>

typedef __attribute__((ext_vector_type(16))) _Float16 v16h;
typedef __attribute__((ext_vector_type(8)))  _Float16 v8h;
typedef __attribute__((ext_vector_type(8)))  float    v8f;
typedef unsigned int u32x4 __attribute__((ext_vector_type(4)));
typedef int          i32x8 __attribute__((ext_vector_type(8)));
typedef int          i32x4 __attribute__((ext_vector_type(4)));

#define NB   32
#define CIN  64
#define COUT 64
#define DH   64
#define NHD  8            // heads
#define TT   128
#define VV   25
#define TV   (TT*VV)      // 3200
#define QKVR (3*NHD*DH)   // 1536
#define HD   (NHD*DH)     // 512
#define KDEEP (DH*TT)     // 8192
#define KCHUNK 512        // scores k-chunk staged via TDM
#define NCHUNK (KDEEP/KCHUNK) // 16
#define CNT_F ((float)(NB*TT*VV)) // 102400

// ---------------- workspace layout (bytes) ----------------
constexpr size_t OFF_XS    = 0;                              // f16 [NB][CIN][TV]
constexpr size_t SZ_XS     = (size_t)NB*CIN*TV*2;
constexpr size_t OFF_WQKV  = OFF_XS + SZ_XS;                 // f16 [1536][64]
constexpr size_t SZ_WQKV   = (size_t)QKVR*CIN*2;
constexpr size_t OFF_WAO   = OFF_WQKV + SZ_WQKV;             // f16 [64][512]
constexpr size_t SZ_WAO    = (size_t)COUT*HD*2;
constexpr size_t OFF_QKV   = OFF_WAO + SZ_WAO;               // f16 [NB][1536][TV]
constexpr size_t SZ_QKV    = (size_t)NB*QKVR*TV*2;
constexpr size_t OFF_PROBS = OFF_QKV + SZ_QKV;               // f16 [NB*NHD][32][32] (transposed, padded)
constexpr size_t SZ_PROBS  = (size_t)NB*NHD*32*32*2;
constexpr size_t OFF_ATT   = OFF_PROBS + SZ_PROBS;           // f16 [NB][512][TV]
constexpr size_t SZ_ATT    = (size_t)NB*HD*TV*2;
constexpr size_t OFF_AO    = OFF_ATT + SZ_ATT;               // f32 [NB][64][TV]
constexpr size_t SZ_F32T   = (size_t)NB*COUT*TV*4;
constexpr size_t OFF_PRE1  = OFF_AO   + SZ_F32T;
constexpr size_t OFF_XS2   = OFF_PRE1 + SZ_F32T;
constexpr size_t OFF_PRE2  = OFF_XS2  + SZ_F32T;
constexpr size_t OFF_STATS = OFF_PRE2 + SZ_F32T;             // stats_s[128], stats_t[128], bnp_s[128], bnp_t[128]

__device__ __forceinline__ int a_kmap(int i, int half) { return ((i < 8) ? i : i + 8) + half * 8; }
__device__ __forceinline__ int b_kmap(int i, int half) { return ((i < 8) ? i : i + 8) + half * 8; }

// ---------------- TDM: 2D tile (f16) global -> LDS ----------------
// tile_rows x tile_cols, global row stride `row_stride` (elements), columns
// beyond `avail_cols` are zero-filled by TDM OOB handling (our 25->32 pad).
#if __has_builtin(__builtin_amdgcn_tensor_load_to_lds)
#define HAVE_TDM 1
#else
#define HAVE_TDM 0
#endif

__device__ __forceinline__ void tdm_load_2d(int lane, _Float16* lds, const _Float16* g,
                                            unsigned avail_cols, unsigned tile_cols,
                                            unsigned tile_rows, unsigned row_stride) {
#if HAVE_TDM
    (void)lane;
    unsigned long long ga = (unsigned long long)(uintptr_t)g;
    unsigned ldsOff = (unsigned)(uintptr_t)(void*)lds;
    u32x4 g0;
    g0[0] = 1u;                                              // count=1, no gather
    g0[1] = ldsOff;                                          // lds_addr [63:32]
    g0[2] = (unsigned)(ga & 0xffffffffu);                    // global_addr [95:64]
    g0[3] = (unsigned)((ga >> 32) & 0x01ffffffu) | (2u << 30); // addr hi + type=2
    i32x8 g1;
    g1[0] = (int)(1u << 16);                                 // data_size=1 (2 bytes)
    g1[1] = (int)((avail_cols & 0xffffu) << 16);             // tensor_dim0 lo16 @bits63:48
    g1[2] = (int)(((avail_cols >> 16) & 0xffffu) |           // tensor_dim0 hi16
                  ((tile_rows & 0xffffu) << 16));            // tensor_dim1 lo16
    g1[3] = (int)(((tile_rows >> 16) & 0xffffu) |            // tensor_dim1 hi16
                  ((tile_cols & 0xffffu) << 16));            // tile_dim0 @bits127:112
    g1[4] = (int)(tile_rows & 0xffffu);                      // tile_dim1; tile_dim2=0
    g1[5] = (int)row_stride;                                 // tensor_dim0_stride lo32
    g1[6] = 0;
    g1[7] = 0;
    i32x4 gz4 = {0, 0, 0, 0};
    i32x8 gz8 = {0, 0, 0, 0, 0, 0, 0, 0};
    // 6-arg toolchain signature: (g0, g1, g2, g3, g4, cpol)
    __builtin_amdgcn_tensor_load_to_lds(g0, g1, gz4, gz4, gz8, 0);
#else
    for (unsigned r = (unsigned)lane; r < tile_rows; r += 32)
        for (unsigned x = 0; x < tile_cols; ++x)
            lds[r * tile_cols + x] =
                (x < avail_cols) ? g[(size_t)r * row_stride + x] : (_Float16)0.0f;
#endif
}

// Wait until at most `N` TDM ops outstanding (in-order completion -> the
// oldest ops have landed in LDS). N must be a compile-time constant (SIMM16).
template <int N>
__device__ __forceinline__ void tdm_wait() {
#if HAVE_TDM
    __builtin_amdgcn_s_wait_tensorcnt(N);
#else
    __syncthreads();
#endif
    asm volatile("" ::: "memory");
}

__device__ __forceinline__ void lds_reuse_fence() {
    asm volatile("s_wait_dscnt 0x0" ::: "memory");
}

__device__ __forceinline__ v16h load_a_rowmajor(const _Float16* Arow, int kk, int half) {
    // A fragment per lane = two contiguous 8-half runs -> two b128 loads
    v8h lo = *(const v8h*)(Arow + kk + half * 8);
    v8h hi = *(const v8h*)(Arow + kk + 16 + half * 8);
    return __builtin_shufflevector(lo, hi, 0, 1, 2, 3, 4, 5, 6, 7,
                                   8, 9, 10, 11, 12, 13, 14, 15);
}

// ---------------- xs = x + positional embedding, cast f16 ----------------
__global__ void k_prep_xs(const float* __restrict__ x, _Float16* __restrict__ xs) {
    size_t idx = (size_t)blockIdx.x * blockDim.x + threadIdx.x;
    if (idx >= (size_t)NB * CIN * TV) return;
    size_t rem = idx % ((size_t)CIN * TV);
    int c  = (int)(rem / TV);
    int tv = (int)(rem % TV);
    float f  = expf(((float)((c >> 1) << 1)) * (-logf(10000.0f) / (float)CIN));
    float ang = (float)tv * f;
    float pe = (c & 1) ? cosf(ang) : sinf(ang);
    xs[idx] = (_Float16)(x[idx] + pe);
}

__global__ void k_cvt_f16(const float* __restrict__ src, _Float16* __restrict__ dst, int n) {
    int i = blockIdx.x * blockDim.x + threadIdx.x;
    if (i < n) dst[i] = (_Float16)src[i];
}

// ---------------- QKV projection: (1536x64) x (64x3200) per sample ----------------
// One TDM stages a K=64 x 64-col B panel; wave computes 4 adjacent 16x16 tiles,
// reusing its A fragments 4x.
__global__ void k_gemm_qkv(const _Float16* __restrict__ W, const _Float16* __restrict__ X,
                           const float* __restrict__ bias, _Float16* __restrict__ out) {
    __shared__ _Float16 sB[CIN * 64];   // 8 KB
    const int lane = threadIdx.x & 31, half = lane >> 4, lm = lane & 15;
    const int c0 = blockIdx.x * 64, m0 = blockIdx.y * 16, n = blockIdx.z;
    tdm_load_2d(lane, sB, X + (size_t)n * CIN * TV + c0, TV - c0, 64, CIN, TV);
    const _Float16* Arow = W + (size_t)(m0 + lm) * CIN;
    v16h a0 = load_a_rowmajor(Arow, 0, half);
    v16h a1 = load_a_rowmajor(Arow, 32, half);
    tdm_wait<0>();
    #pragma unroll
    for (int j = 0; j < 4; ++j) {
        v8f acc = {};
        v16h b;
        #pragma unroll
        for (int i = 0; i < 16; ++i)
            b[i] = sB[(0 + b_kmap(i, half)) * 64 + j * 16 + lm];
        acc = __builtin_amdgcn_wmma_f32_16x16x32_f16(false, a0, false, b, (short)0, acc, false, false);
        #pragma unroll
        for (int i = 0; i < 16; ++i)
            b[i] = sB[(32 + b_kmap(i, half)) * 64 + j * 16 + lm];
        acc = __builtin_amdgcn_wmma_f32_16x16x32_f16(false, a1, false, b, (short)0, acc, false, false);
        #pragma unroll
        for (int r = 0; r < 8; ++r) {
            int o = m0 + r + half * 8;
            out[(size_t)n * QKVR * TV + (size_t)o * TV + c0 + j * 16 + lm] =
                (_Float16)(acc[r] + bias[o]);
        }
    }
}

// ---------------- scores (K=8192) + tanh gate -> probs^T (padded 32x32) ----------------
// Double-buffered TDM pipeline: DMA of chunk c+1 overlaps the 16 WMMAs of
// chunk c (in-order TENSORcnt: wait<=2 means the current chunk's 2 panels
// have landed). TDM OOB zero-fill does the 25->32 pad.
__global__ void k_scores(const _Float16* __restrict__ qkv, const float* __restrict__ alphas,
                         const float* __restrict__ att0s, _Float16* __restrict__ probsT) {
    __shared__ _Float16 sA[2][KCHUNK * 16];  // 2 x 16 KB
    __shared__ _Float16 sB[2][KCHUNK * 16];  // 2 x 16 KB
    const int lane = threadIdx.x & 31, half = lane >> 4, lm = lane & 15;
    const int ut = blockIdx.x >> 1, vt = blockIdx.x & 1;
    const int nh = blockIdx.y, n = nh / NHD, h = nh % NHD;
    const _Float16* qm = qkv + (size_t)n * QKVR * TV + (size_t)(0 * HD + h * DH) * TV; // (8192 x 25)
    const _Float16* km = qkv + (size_t)n * QKVR * TV + (size_t)(1 * HD + h * DH) * TV; // (8192 x 25)
    const int u0 = ut * 16, v0 = vt * 16;
    v8f acc = {};
    tdm_load_2d(lane, sA[0], km + u0, VV - u0, 16, KCHUNK, VV);
    tdm_load_2d(lane, sB[0], qm + v0, VV - v0, 16, KCHUNK, VV);
    for (int c = 0; c < NCHUNK; ++c) {
        const int cur = c & 1;
        if (c + 1 < NCHUNK) {   // prefetch next chunk into the other buffer
            size_t koff = (size_t)(c + 1) * KCHUNK * VV;
            tdm_load_2d(lane, sA[cur ^ 1], km + koff + u0, VV - u0, 16, KCHUNK, VV);
            tdm_load_2d(lane, sB[cur ^ 1], qm + koff + v0, VV - v0, 16, KCHUNK, VV);
            tdm_wait<2>();      // current chunk's panels are in LDS, next in flight
        } else {
            tdm_wait<0>();
        }
        const _Float16* pa = sA[cur];
        const _Float16* pb = sB[cur];
        for (int ko = 0; ko < KCHUNK; ko += 32) {
            v16h a, b;
            #pragma unroll
            for (int i = 0; i < 16; ++i)
                a[i] = pa[(ko + a_kmap(i, half)) * 16 + lm];
            #pragma unroll
            for (int i = 0; i < 16; ++i)
                b[i] = pb[(ko + b_kmap(i, half)) * 16 + lm];
            acc = __builtin_amdgcn_wmma_f32_16x16x32_f16(false, a, false, b, (short)0, acc, false, false);
        }
        lds_reuse_fence();  // LDS reads done before this buffer is overwritten (c+2)
    }
    const float alpha = alphas[h];
    const float scale = (float)TT / 8.0f;   // T / sqrt(D) = 16
    #pragma unroll
    for (int r = 0; r < 8; ++r) {
        int u = u0 + r + half * 8, v = v0 + lm;
        float p = 0.0f;
        if (u < VV && v < VV)
            p = tanhf(acc[r] * scale) * alpha + att0s[(size_t)h * VV * VV + u * VV + v];
        probsT[(size_t)nh * 1024 + v * 32 + u] = (_Float16)p;  // transposed store
    }
}

// ---------------- att = V(8192x25) x probs^T(25x25), K padded to 32 ----------------
__global__ void k_att(const _Float16* __restrict__ qkv, const _Float16* __restrict__ probsT,
                      _Float16* __restrict__ att) {
    const int lane = threadIdx.x & 31, half = lane >> 4, lm = lane & 15;
    const int m0 = blockIdx.x * 16, u0 = blockIdx.y * 16;
    const int nh = blockIdx.z, n = nh / NHD, h = nh % NHD;
    const _Float16* vm = qkv + (size_t)n * QKVR * TV + (size_t)(2 * HD + h * DH) * TV; // (8192 x 25)
    const _Float16* pT = probsT + (size_t)nh * 1024;                                   // [v][u] 32x32
    v16h a, b;
    #pragma unroll
    for (int i = 0; i < 16; ++i) {
        int v = a_kmap(i, half);
        a[i] = (v < VV) ? vm[(size_t)(m0 + lm) * VV + v] : (_Float16)0.0f;
    }
    #pragma unroll
    for (int i = 0; i < 16; ++i) {
        int v = b_kmap(i, half);
        b[i] = pT[v * 32 + u0 + lm];
    }
    v8f acc = {};
    acc = __builtin_amdgcn_wmma_f32_16x16x32_f16(false, a, false, b, (short)0, acc, false, false);
    #pragma unroll
    for (int r = 0; r < 8; ++r) {
        int m = m0 + r + half * 8, u = u0 + lm;
        if (u < VV)
            att[(size_t)n * HD * TV + (size_t)h * DH * TV + (size_t)m * VV + u] = (_Float16)acc[r];
    }
}

// ---------------- attn_out: (64x512) x (512x3200) per sample, f32 out ----------------
// One TDM stages a K=512 x 64-col panel (64 KB LDS); wave computes 4 tiles.
__global__ void k_gemm_ao(const _Float16* __restrict__ W, const _Float16* __restrict__ X,
                          const float* __restrict__ bias, float* __restrict__ out) {
    __shared__ _Float16 sB[HD * 64];    // 64 KB
    const int lane = threadIdx.x & 31, half = lane >> 4, lm = lane & 15;
    const int c0 = blockIdx.x * 64, m0 = blockIdx.y * 16, n = blockIdx.z;
    tdm_load_2d(lane, sB, X + (size_t)n * HD * TV + c0, TV - c0, 64, HD, TV);
    const _Float16* Arow = W + (size_t)(m0 + lm) * HD;
    tdm_wait<0>();
    #pragma unroll
    for (int j = 0; j < 4; ++j) {
        v8f acc = {};
        for (int kk = 0; kk < HD; kk += 32) {
            v16h a = load_a_rowmajor(Arow, kk, half);
            v16h b;
            #pragma unroll
            for (int i = 0; i < 16; ++i)
                b[i] = sB[(kk + b_kmap(i, half)) * 64 + j * 16 + lm];
            acc = __builtin_amdgcn_wmma_f32_16x16x32_f16(false, a, false, b, (short)0, acc, false, false);
        }
        #pragma unroll
        for (int r = 0; r < 8; ++r) {
            int o = m0 + r + half * 8;
            out[(size_t)n * COUT * TV + (size_t)o * TV + c0 + j * 16 + lm] = acc[r] + bias[o];
        }
    }
}

// ---------------- spatial conv (1x3 along V) + per-channel stats ----------------
__global__ void k_conv_s(const float* __restrict__ in, const float* __restrict__ W,
                         const float* __restrict__ bias, float* __restrict__ out,
                         float* __restrict__ stats) {
    const int n = blockIdx.x / COUT, o = blockIdx.x % COUT;
    __shared__ float w[CIN * 3];
    __shared__ float s_sum[256], s_sq[256];
    for (int j = threadIdx.x; j < CIN * 3; j += blockDim.x) w[j] = W[(size_t)o * CIN * 3 + j];
    __syncthreads();
    const float* inb = in + (size_t)n * COUT * TV;
    float lsum = 0.0f, lsq = 0.0f;
    for (int e = threadIdx.x; e < TV; e += blockDim.x) {
        int t = e / VV, v = e % VV;
        float acc = bias[o];
        for (int ci = 0; ci < CIN; ++ci) {
            const float* row = inb + (size_t)ci * TV + t * VV;
            const float* wr = w + ci * 3;
            if (v - 1 >= 0) acc = fmaf(wr[0], row[v - 1], acc);
            acc = fmaf(wr[1], row[v], acc);
            if (v + 1 < VV) acc = fmaf(wr[2], row[v + 1], acc);
        }
        out[(size_t)n * COUT * TV + (size_t)o * TV + e] = acc;
        lsum += acc; lsq = fmaf(acc, acc, lsq);
    }
    s_sum[threadIdx.x] = lsum; s_sq[threadIdx.x] = lsq;
    __syncthreads();
    for (int s = 128; s > 0; s >>= 1) {
        if (threadIdx.x < s) {
            s_sum[threadIdx.x] += s_sum[threadIdx.x + s];
            s_sq[threadIdx.x]  += s_sq[threadIdx.x + s];
        }
        __syncthreads();
    }
    if (threadIdx.x == 0) {
        atomicAdd(&stats[o], s_sum[0]);
        atomicAdd(&stats[COUT + o], s_sq[0]);
    }
}

// ---------------- time conv (5x1 along T) + per-channel stats ----------------
__global__ void k_conv_t(const float* __restrict__ in, const float* __restrict__ W,
                         const float* __restrict__ bias, float* __restrict__ out,
                         float* __restrict__ stats) {
    const int n = blockIdx.x / COUT, o = blockIdx.x % COUT;
    __shared__ float w[CIN * 5];
    __shared__ float s_sum[256], s_sq[256];
    for (int j = threadIdx.x; j < CIN * 5; j += blockDim.x) w[j] = W[(size_t)o * CIN * 5 + j];
    __syncthreads();
    const float* inb = in + (size_t)n * COUT * TV;
    float lsum = 0.0f, lsq = 0.0f;
    for (int e = threadIdx.x; e < TV; e += blockDim.x) {
        int t = e / VV, v = e % VV;
        float acc = bias[o];
        for (int ci = 0; ci < CIN; ++ci) {
            const float* col = inb + (size_t)ci * TV + v;
            const float* wr = w + ci * 5;
            #pragma unroll
            for (int kt = 0; kt < 5; ++kt) {
                int tt = t + kt - 2;
                if (tt >= 0 && tt < TT) acc = fmaf(wr[kt], col[tt * VV], acc);
            }
        }
        out[(size_t)n * COUT * TV + (size_t)o * TV + e] = acc;
        lsum += acc; lsq = fmaf(acc, acc, lsq);
    }
    s_sum[threadIdx.x] = lsum; s_sq[threadIdx.x] = lsq;
    __syncthreads();
    for (int s = 128; s > 0; s >>= 1) {
        if (threadIdx.x < s) {
            s_sum[threadIdx.x] += s_sum[threadIdx.x + s];
            s_sq[threadIdx.x]  += s_sq[threadIdx.x + s];
        }
        __syncthreads();
    }
    if (threadIdx.x == 0) {
        atomicAdd(&stats[o], s_sum[0]);
        atomicAdd(&stats[COUT + o], s_sq[0]);
    }
}

// ---------------- bn: stats -> scale/shift ----------------
__global__ void k_bn_fin(const float* __restrict__ stats, const float* __restrict__ gamma,
                         const float* __restrict__ beta, float* __restrict__ bnp) {
    int o = threadIdx.x;
    if (o < COUT) {
        float mean = stats[o] / CNT_F;
        float var  = stats[COUT + o] / CNT_F - mean * mean;
        float sc   = gamma[o] * rsqrtf(var + 1e-5f);
        bnp[o] = sc;
        bnp[COUT + o] = beta[o] - mean * sc;
    }
}

// ---------------- y = relu(pre*scale + shift + residual) ----------------
__global__ void k_bn_apply(const float* __restrict__ pre, const float* __restrict__ res,
                           const float* __restrict__ bnp, float* __restrict__ out) {
    size_t idx = (size_t)blockIdx.x * blockDim.x + threadIdx.x;
    if (idx >= (size_t)NB * COUT * TV) return;
    int o = (int)((idx / TV) % COUT);
    float y = fmaf(pre[idx], bnp[o], bnp[COUT + o]) + res[idx];
    out[idx] = fmaxf(y, 0.0f);
}

extern "C" void kernel_launch(void* const* d_in, const int* in_sizes, int n_in,
                              void* d_out, int out_size, void* d_ws, size_t ws_size,
                              hipStream_t stream) {
    (void)in_sizes; (void)n_in; (void)out_size; (void)ws_size;
    const float* x          = (const float*)d_in[0];
    const float* W_qkv      = (const float*)d_in[1];
    const float* b_qkv      = (const float*)d_in[2];
    const float* alphas     = (const float*)d_in[3];
    const float* att0s      = (const float*)d_in[4];
    const float* W_attn_out = (const float*)d_in[5];
    const float* b_attn_out = (const float*)d_in[6];
    const float* W_spatial  = (const float*)d_in[7];
    const float* b_spatial  = (const float*)d_in[8];
    const float* W_time     = (const float*)d_in[9];
    const float* b_time     = (const float*)d_in[10];
    const float* gamma      = (const float*)d_in[11];
    const float* beta       = (const float*)d_in[12];
    float* out = (float*)d_out;

    char* ws = (char*)d_ws;
    _Float16* xs_f16   = (_Float16*)(ws + OFF_XS);
    _Float16* wqkv_f16 = (_Float16*)(ws + OFF_WQKV);
    _Float16* wao_f16  = (_Float16*)(ws + OFF_WAO);
    _Float16* qkv_f16  = (_Float16*)(ws + OFF_QKV);
    _Float16* probsT   = (_Float16*)(ws + OFF_PROBS);
    _Float16* att_f16  = (_Float16*)(ws + OFF_ATT);
    float*    att_out  = (float*)(ws + OFF_AO);
    float*    pre1     = (float*)(ws + OFF_PRE1);
    float*    xs2      = (float*)(ws + OFF_XS2);
    float*    pre2     = (float*)(ws + OFF_PRE2);
    float*    stats_s  = (float*)(ws + OFF_STATS);
    float*    stats_t  = stats_s + 2 * COUT;
    float*    bnp_s    = stats_t + 2 * COUT;
    float*    bnp_t    = bnp_s + 2 * COUT;

    (void)hipMemsetAsync(stats_s, 0, 4 * COUT * sizeof(float), stream);  // stats_s + stats_t

    const size_t total = (size_t)NB * COUT * TV;  // 6,553,600
    const int TB = 256;
    const int nb_total = (int)((total + TB - 1) / TB);

    // 1) xs = x + pos_embed, f16 ; weight conversions
    k_prep_xs<<<nb_total, TB, 0, stream>>>(x, xs_f16);
    k_cvt_f16<<<(QKVR * CIN + TB - 1) / TB, TB, 0, stream>>>(W_qkv, wqkv_f16, QKVR * CIN);
    k_cvt_f16<<<(COUT * HD + TB - 1) / TB, TB, 0, stream>>>(W_attn_out, wao_f16, COUT * HD);

    // 2) QKV GEMM (WMMA; one TDM per 64-col panel, 4 tiles per wave)
    k_gemm_qkv<<<dim3(TV / 64, QKVR / 16, NB), 32, 0, stream>>>(wqkv_f16, xs_f16, b_qkv, qkv_f16);

    // 3) scores + tanh gate (WMMA, K=8192, double-buffered TDM pipeline)
    k_scores<<<dim3(4, NB * NHD), 32, 0, stream>>>(qkv_f16, alphas, att0s, probsT);

    // 4) att = V x probs^T (WMMA)
    k_att<<<dim3(KDEEP / 16, 2, NB * NHD), 32, 0, stream>>>(qkv_f16, probsT, att_f16);

    // 5) attention output projection (WMMA; one TDM per 64-col panel, K=512)
    k_gemm_ao<<<dim3(TV / 64, COUT / 16, NB), 32, 0, stream>>>(wao_f16, att_f16, b_attn_out, att_out);

    // 6) spatial conv + stats, bn finalize, apply (+residual, relu)
    k_conv_s<<<NB * COUT, 256, 0, stream>>>(att_out, W_spatial, b_spatial, pre1, stats_s);
    k_bn_fin<<<1, 64, 0, stream>>>(stats_s, gamma, beta, bnp_s);
    k_bn_apply<<<nb_total, TB, 0, stream>>>(pre1, x, bnp_s, xs2);

    // 7) temporal conv + stats, bn finalize, apply (+residual, relu) -> output
    k_conv_t<<<NB * COUT, 256, 0, stream>>>(xs2, W_time, b_time, pre2, stats_t);
    k_bn_fin<<<1, 64, 0, stream>>>(stats_t, gamma, beta, bnp_t);
    k_bn_apply<<<nb_total, TB, 0, stream>>>(pre2, x, bnp_t, out);
}
